// DeformableBlock_19954418057774
// MI455X (gfx1250) — matrix operand order repaired and checked by reading, compiled, not run on hardware
//
#include <hip/hip_runtime.h>
#include <hip/hip_bf16.h>
#include <math.h>

typedef __attribute__((ext_vector_type(16))) _Float16 v16h;
typedef __attribute__((ext_vector_type(8)))  float    v8f;

#define DEV __device__ __forceinline__

// ---- WMMA fragment helpers (CDNA5 16x16x32 f16 layouts, wave32) ----

// A matrix 16x32 (MxK) from row-major f16 buffer (LDS): lane m=lane&15,
// half kh=lane>>4; VGPR i holds K = (i&4)*4 + kh*8 + (i&3)*2 (+1).
DEV v16h fragA_lds(const _Float16* base, int ld, int row0, int k0) {
  const int lane = threadIdx.x & 31;
  const int m = lane & 15, kh = lane >> 4;
  v16h a;
  const _Float16* rp = base + (size_t)(row0 + m) * ld + k0;
#pragma unroll
  for (int i = 0; i < 8; ++i) {
    int kb = ((i & 4) << 2) + (kh << 3) + ((i & 3) << 1);
    a[2 * i]     = rp[kb];
    a[2 * i + 1] = rp[kb + 1];
  }
  return a;
}

// B matrix 32x16 (KxN) from row-major f32 global weights, converted to f16.
// lane n=lane&15, kh=lane>>4; VGPR i holds K = (i&4)*4 + kh*8 + (i&3)*2 (+1).
// Prefetches the next K-slab of the weight stream (global_prefetch_b8).
DEV v16h fragB_g32(const float* W, int ld, int k0, int n0) {
  const int lane = threadIdx.x & 31;
  const int n = lane & 15, kh = lane >> 4;
  v16h b;
  __builtin_prefetch(W + (size_t)(k0 + 32) * ld + n0 + n, 0, 1);
#pragma unroll
  for (int i = 0; i < 8; ++i) {
    int kb = ((i & 4) << 2) + (kh << 3) + ((i & 3) << 1);
    b[2 * i]     = (_Float16)W[(size_t)(k0 + kb) * ld + n0 + n];
    b[2 * i + 1] = (_Float16)W[(size_t)(k0 + kb + 1) * ld + n0 + n];
  }
  return b;
}

DEV v8f wmma32(v16h a, v16h b, v8f c) {
  return __builtin_amdgcn_wmma_f32_16x16x32_f16(
      /*neg_a=*/false, a, /*neg_b=*/false, b,
      /*c_mod=*/(short)0, c, /*reuse_a=*/false, /*reuse_b=*/false);
}

DEV float wred_sum(float v) {
#pragma unroll
  for (int m = 16; m >= 1; m >>= 1) v += __shfl_xor(v, m, 32);
  return v;
}

// Branch-free tanh: 1 - 2/(e^{2|x|}+1), sign restored. No EXEC divergence.
DEV float fast_tanh(float x) {
  float e = __expf(2.f * fabsf(x));
  return copysignf(1.f - 2.f / (e + 1.f), x);
}

// tanh-form GELU (branch-free)
DEV float fast_gelu(float x) {
  float u = 0.7978845608f * (x + 0.044715f * x * x * x);
  return 0.5f * x * (1.f + fast_tanh(u));
}

// =====================================================================
// Kernel A: h = LN(xq + x0); logits = h@aw_W (+softmax over NS=4),
//           offsets = tanh(h@so_W) + ref  -> pos_ws, w_ws
// grid (8, 5, 4) = (token tile of 128, level, batch), 256 threads
// =====================================================================
__global__ __launch_bounds__(256) void kA(
    const float* __restrict__ x, const float* __restrict__ ref,
    const float* __restrict__ g1, const float* __restrict__ b1,
    const float* __restrict__ awW, const float* __restrict__ awb,
    const float* __restrict__ soW, const float* __restrict__ sob,
    float* __restrict__ pos_ws, float* __restrict__ w_ws) {
  __shared__ _Float16 hbuf[128 * 256];
  const int tid = threadIdx.x, lane = tid & 31, wave = tid >> 5;
  const int t0 = blockIdx.x * 128;
  const int l  = blockIdx.y;
  const int bb = blockIdx.z;
  const float* xq = x + (size_t)(bb * 6 + l + 1) * 1024 * 256;
  const float* x0 = x + (size_t)(bb * 6) * 1024 * 256;

  // ---- LayerNorm: each wave handles 16 tokens, 8 channels per lane ----
  for (int tt = wave; tt < 128; tt += 8) {
    const int tok = t0 + tt;
    float v[8], s = 0.f, sq = 0.f;
#pragma unroll
    for (int j = 0; j < 8; ++j) {
      int c = lane * 8 + j;
      v[j] = xq[(size_t)tok * 256 + c] + x0[(size_t)tok * 256 + c];
      s += v[j]; sq += v[j] * v[j];
    }
    s = wred_sum(s); sq = wred_sum(sq);
    float mean = s * (1.f / 256.f);
    float rstd = rsqrtf(sq * (1.f / 256.f) - mean * mean + 1e-5f);
#pragma unroll
    for (int j = 0; j < 8; ++j) {
      int c = lane * 8 + j;
      hbuf[tt * 256 + c] = (_Float16)((v[j] - mean) * rstd * g1[c] + b1[c]);
    }
  }
  __syncthreads();

  const int mt = wave;
  const int kh = lane >> 4, n = lane & 15;
  const size_t tgb = (size_t)(bb * 5 + l) * 1024;

  // ---- attention-weight GEMM 128x256 @ 256x32, softmax over NS=4 ----
  for (int nt = 0; nt < 2; ++nt) {
    v8f acc = {};
#pragma unroll
    for (int ks = 0; ks < 8; ++ks) {
      acc = wmma32(fragA_lds(hbuf, 256, mt * 16, ks * 32),
                   fragB_g32(awW, 32, ks * 32, nt * 16), acc);
    }
    int j = nt * 16 + n;
    float bias = awb[j];
#pragma unroll
    for (int r = 0; r < 8; ++r) {
      float e = acc[r] + bias;
      float mx = fmaxf(e, __shfl_xor(e, 1, 32));
      mx = fmaxf(mx, __shfl_xor(mx, 2, 32));
      float ex = __expf(e - mx);
      float sm = ex + __shfl_xor(ex, 1, 32);
      sm += __shfl_xor(sm, 2, 32);
      int tok = t0 + mt * 16 + r + 8 * kh;
      w_ws[(tgb + tok) * 32 + j] = ex / sm;
    }
  }

  // ---- sampling-offset GEMM 128x256 @ 256x64, tanh + ref ----
  for (int nt = 0; nt < 4; ++nt) {
    v8f acc = {};
#pragma unroll
    for (int ks = 0; ks < 8; ++ks) {
      acc = wmma32(fragA_lds(hbuf, 256, mt * 16, ks * 32),
                   fragB_g32(soW, 64, ks * 32, nt * 16), acc);
    }
    int j = nt * 16 + n;
    float bias = sob[j];
    int coord = j & 1;
#pragma unroll
    for (int r = 0; r < 8; ++r) {
      int tok = t0 + mt * 16 + r + 8 * kh;
      float p = fast_tanh(acc[r] + bias) +
                ref[((size_t)bb * 1024 + tok) * 2 + coord];
      pos_ws[(tgb + tok) * 64 + j] = p;
    }
  }
}

// =====================================================================
// Kernel B (one launch per level): bilinear-gather + weighted combine
// over NS into (16 tok * 8 head) x C_l f16 tile, then WMMA-project by
// ep_W (C_l x 32), add residual -> xq2_ws. grid (64, 4), 256 threads.
// =====================================================================
__global__ __launch_bounds__(256) void kB(
    const float* __restrict__ feat, int C, int HW, int shiftC,
    const float* __restrict__ epW, const float* __restrict__ epb,
    const float* __restrict__ x, int l,
    const float* __restrict__ pos_ws, const float* __restrict__ w_ws,
    float* __restrict__ xq2_ws) {
  extern __shared__ char smem[];
  _Float16* comb = (_Float16*)smem;                         // 128 * C
  float* swx  = (float*)(smem + (size_t)128 * C * sizeof(_Float16));
  float* swy  = swx + 512;
  float* swgt = swy + 512;
  int* sx0 = (int*)(swgt + 512);
  int* sy0 = sx0 + 512;
  int* sx1 = sy0 + 512;
  int* sy1 = sx1 + 512;

  const int tid = threadIdx.x, lane = tid & 31, wave = tid >> 5;
  const int bb = blockIdx.y;
  const int t0 = blockIdx.x * 16;
  const size_t tgb = (size_t)(bb * 5 + l) * 1024 + t0;

  // ---- precompute bilinear coords + softmax weights for 16*32 samples ----
  for (int e = tid; e < 512; e += 256) {
    int tok = e >> 5, pt = e & 31;
    size_t tg = tgb + tok;
    float px = pos_ws[tg * 64 + pt * 2 + 0];
    float py = pos_ws[tg * 64 + pt * 2 + 1];
    float lim = (float)(HW - 1);
    float gx = fminf(fmaxf((px + 1.f) * 0.5f * lim, 0.f), lim);
    float gy = fminf(fmaxf((py + 1.f) * 0.5f * lim, 0.f), lim);
    float fx = floorf(gx), fy = floorf(gy);
    swx[e] = gx - fx; swy[e] = gy - fy;
    int x0i = (int)fx, y0i = (int)fy;
    sx0[e] = x0i; sy0[e] = y0i;
    sx1[e] = min(x0i + 1, HW - 1); sy1[e] = min(y0i + 1, HW - 1);
    swgt[e] = w_ws[tg * 32 + pt];
  }
  __syncthreads();

  // ---- gather + combine over s (linear commute of weight & projection) ----
  const int total = 128 << shiftC;  // 128 rows * C cols
  for (int idx = tid; idx < total; idx += 256) {
    int row = idx >> shiftC;
    int c = idx & (C - 1);
    int tok = row >> 3, head = row & 7;
    const float* fb = feat + (size_t)(bb * C + c) * HW * HW;
    float acc = 0.f;
#pragma unroll
    for (int s = 0; s < 4; ++s) {
      int e = tok * 32 + head * 4 + s;
      int x0i = sx0[e], x1i = sx1[e], y0i = sy0[e], y1i = sy1[e];
      float wx = swx[e], wy = swy[e];
      float v00 = fb[(size_t)y0i * HW + x0i];
      float v01 = fb[(size_t)y0i * HW + x1i];
      float v10 = fb[(size_t)y1i * HW + x0i];
      float v11 = fb[(size_t)y1i * HW + x1i];
      float bil = v00 * (1.f - wx) * (1.f - wy) + v01 * wx * (1.f - wy) +
                  v10 * (1.f - wx) * wy + v11 * wx * wy;
      acc += swgt[e] * bil;
    }
    comb[(size_t)row * C + c] = (_Float16)acc;
  }
  __syncthreads();

  // ---- WMMA: (128 x C) @ (C x 32) + bias + residual -> xq2 ----
  const int kh = lane >> 4, n = lane & 15;
  const int ksteps = C >> 5;
  for (int nt = 0; nt < 2; ++nt) {
    v8f acc = {};
#pragma unroll 4
    for (int ks = 0; ks < ksteps; ++ks) {
      acc = wmma32(fragA_lds(comb, C, wave * 16, ks * 32),
                   fragB_g32(epW, 32, ks * 32, nt * 16), acc);
    }
    int d = nt * 16 + n;
    float bias = epb[d];
#pragma unroll
    for (int r = 0; r < 8; ++r) {
      int row = wave * 16 + r + 8 * kh;
      int tok = row >> 3, head = row & 7;
      int ch = head * 32 + d;
      float res = x[((size_t)(bb * 6 + l + 1) * 1024 + t0 + tok) * 256 + ch];
      xq2_ws[(tgb + tok) * 256 + ch] = res + acc[r] + bias;
    }
  }
}

// =====================================================================
// Kernel C: h2 = LN(xq2); mlp = gelu(h2@fc1)@fc2; out = xq2 + mlp
// grid (16, 20) = (64-token tile, b*5+l), 256 threads, dyn LDS 96KB
// =====================================================================
__global__ __launch_bounds__(256) void kC(
    const float* __restrict__ xq2_ws,
    const float* __restrict__ g2, const float* __restrict__ b2,
    const float* __restrict__ fc1W, const float* __restrict__ fc1b,
    const float* __restrict__ fc2W, const float* __restrict__ fc2b,
    float* __restrict__ out) {
  extern __shared__ char smem[];
  _Float16* h2  = (_Float16*)smem;        // 64 x 256
  _Float16* act = h2 + 64 * 256;          // 64 x 512
  const int tid = threadIdx.x, lane = tid & 31, wave = tid >> 5;
  const int bl = blockIdx.y;
  const int bb = bl / 5, l = bl % 5;
  const int t0 = blockIdx.x * 64;
  const size_t tgb = (size_t)bl * 1024 + t0;

  for (int tt = wave; tt < 64; tt += 8) {
    float v[8], s = 0.f, sq = 0.f;
#pragma unroll
    for (int j = 0; j < 8; ++j) {
      int c = lane * 8 + j;
      v[j] = xq2_ws[(tgb + tt) * 256 + c];
      s += v[j]; sq += v[j] * v[j];
    }
    s = wred_sum(s); sq = wred_sum(sq);
    float mean = s * (1.f / 256.f);
    float rstd = rsqrtf(sq * (1.f / 256.f) - mean * mean + 1e-5f);
#pragma unroll
    for (int j = 0; j < 8; ++j) {
      int c = lane * 8 + j;
      h2[tt * 256 + c] = (_Float16)((v[j] - mean) * rstd * g2[c] + b2[c]);
    }
  }
  __syncthreads();

  const int kh = lane >> 4, n = lane & 15;
  // ---- GEMM1 (64x256)@(256x512) + GELU ----
  for (int jj = wave; jj < 128; jj += 8) {
    int mt = jj & 3, nt = jj >> 2;
    v8f acc = {};
#pragma unroll
    for (int ks = 0; ks < 8; ++ks) {
      acc = wmma32(fragA_lds(h2, 256, mt * 16, ks * 32),
                   fragB_g32(fc1W, 512, ks * 32, nt * 16), acc);
    }
    int col = nt * 16 + n;
    float bias = fc1b[col];
#pragma unroll
    for (int r = 0; r < 8; ++r) {
      int row = mt * 16 + r + 8 * kh;
      act[row * 512 + col] = (_Float16)fast_gelu(acc[r] + bias);
    }
  }
  __syncthreads();

  // ---- GEMM2 (64x512)@(512x256) + residual -> out ----
  for (int jj = wave; jj < 64; jj += 8) {
    int mt = jj & 3, nt = jj >> 2;
    v8f acc = {};
#pragma unroll 4
    for (int ks = 0; ks < 16; ++ks) {
      acc = wmma32(fragA_lds(act, 512, mt * 16, ks * 32),
                   fragB_g32(fc2W, 256, ks * 32, nt * 16), acc);
    }
    int col = nt * 16 + n;
    float bias = fc2b[col];
#pragma unroll
    for (int r = 0; r < 8; ++r) {
      int row = mt * 16 + r + 8 * kh;
      float res = xq2_ws[(tgb + row) * 256 + col];
      out[((size_t)(bb * 6 + l + 1) * 1024 + t0 + row) * 256 + col] =
          res + acc[r] + bias;
    }
  }
}

// ---- Kernel D: pass x0_tok through to output slot 0 ----
__global__ __launch_bounds__(256) void kD(const float* __restrict__ x,
                                          float* __restrict__ out) {
  size_t i = (size_t)blockIdx.x * blockDim.x + threadIdx.x;  // float4 idx
  const size_t per_b = (size_t)1024 * 256 / 4;
  size_t bb = i / per_b, rem = i % per_b;
  const float4* src = (const float4*)(x + bb * 6 * 1024 * 256) + rem;
  float4* dst = (float4*)((float*)out + bb * 6 * 1024 * 256) + rem;
  *dst = *src;
}

extern "C" void kernel_launch(void* const* d_in, const int* in_sizes, int n_in,
                              void* d_out, int out_size, void* d_ws,
                              size_t ws_size, hipStream_t stream) {
  (void)in_sizes; (void)n_in; (void)out_size; (void)ws_size;
  const float* x    = (const float*)d_in[0];
  const float* ref  = (const float*)d_in[1];
  const float* feats[5] = {(const float*)d_in[2], (const float*)d_in[3],
                           (const float*)d_in[4], (const float*)d_in[5],
                           (const float*)d_in[6]};
  const float* n1g = (const float*)d_in[7];
  const float* n1b = (const float*)d_in[8];
  const float* awW = (const float*)d_in[9];
  const float* awb = (const float*)d_in[10];
  const float* soW = (const float*)d_in[11];
  const float* sob = (const float*)d_in[12];
  const float* epW[5], *epb[5];
  for (int l = 0; l < 5; ++l) {
    epW[l] = (const float*)d_in[13 + 2 * l];
    epb[l] = (const float*)d_in[14 + 2 * l];
  }
  const float* n2g  = (const float*)d_in[23];
  const float* n2b  = (const float*)d_in[24];
  const float* fc1W = (const float*)d_in[25];
  const float* fc1b = (const float*)d_in[26];
  const float* fc2W = (const float*)d_in[27];
  const float* fc2b = (const float*)d_in[28];

  float* ws     = (float*)d_ws;
  float* pos_ws = ws;                                     // 4*5*1024*64
  float* w_ws   = pos_ws + (size_t)4 * 5 * 1024 * 64;     // 4*5*1024*32
  float* xq2_ws = w_ws + (size_t)4 * 5 * 1024 * 32;       // 4*5*1024*256

  kA<<<dim3(8, 5, 4), 256, 0, stream>>>(x, ref, n1g, n1b, awW, awb, soW, sob,
                                        pos_ws, w_ws);

  const int Cs[5] = {64, 128, 256, 512, 256};
  const int Hs[5] = {128, 64, 32, 16, 16};
  const int sh[5] = {6, 7, 8, 9, 8};
  for (int l = 0; l < 5; ++l) {
    size_t sm = (size_t)128 * Cs[l] * sizeof(_Float16) + 512 * 4 * 7;
    kB<<<dim3(64, 4), 256, sm, stream>>>(feats[l], Cs[l], Hs[l], sh[l],
                                         epW[l], epb[l], x, l, pos_ws, w_ws,
                                         xq2_ws);
  }

  kC<<<dim3(16, 20), 256, 64 * 256 * 2 + 64 * 512 * 2, stream>>>(
      xq2_ws, n2g, n2b, fc1W, fc1b, fc2W, fc2b, (float*)d_out);

  kD<<<dim3(1024), 256, 0, stream>>>(x, (float*)d_out);
}
